// Encoder_88313117540563
// MI455X (gfx1250) — compile-verified, matrix operand order checked by reading, not run on hardware
//
#include <hip/hip_runtime.h>

typedef _Float16 v16h __attribute__((ext_vector_type(16)));
typedef _Float16 v8h  __attribute__((ext_vector_type(8)));
typedef float    v8f  __attribute__((ext_vector_type(8)));

#define T_STEPS 4096
#define BATCH   512
#define H1      64
#define H2      32
#define LAT     16

#define H1LD 72   // padded f16 row stride for h1 (multiple of 8 -> 16B aligned rows)
#define H2LD 40   // padded f16 row stride for h2

#define LOG2E 1.442695040888963f

// Fast hardware transcendentals: v_exp_f32 / v_rcp_f32 (TRANS ops, co-execute with XDL).
__device__ __forceinline__ float fexp(float x) {
    return __builtin_amdgcn_exp2f(x * LOG2E);
}
__device__ __forceinline__ float frcp(float x) {
    return __builtin_amdgcn_rcpf(x);
}
__device__ __forceinline__ float sigmoidf_fast(float x) {
    return frcp(1.0f + fexp(-x));
}
__device__ __forceinline__ float tanhf_fast(float x) {
#if __has_builtin(__builtin_amdgcn_tanhf)
    return __builtin_amdgcn_tanhf(x);     // gfx1250 v_tanh_f32
#else
    return 1.0f - 2.0f * frcp(fexp(2.0f * x) + 1.0f);
#endif
}

// Build a WMMA B fragment (K=32 chunk) for gates = h @ W^T, i.e. B[k][n] = W[col_n][k].
// W is row-major f32, row length `ld`. Lane n (n = lane&15) owns output column `col`.
// K packing per 16-bit WMMA layout: lanes 0-15 hold k = k0+{0..7,16..23},
// lanes 16-31 hold k = k0+{8..15,24..31}.
__device__ __forceinline__ v16h load_bfrag_f32(const float* __restrict__ W, int ld,
                                               int col, int k0, int lane) {
    const int khalf = (lane & 16) ? 8 : 0;
    const float* p = W + (size_t)col * ld + k0 + khalf;
    float4 a = *(const float4*)(p);
    float4 b = *(const float4*)(p + 4);
    float4 c = *(const float4*)(p + 16);
    float4 d = *(const float4*)(p + 20);
    float t[16] = {a.x,a.y,a.z,a.w, b.x,b.y,b.z,b.w,
                   c.x,c.y,c.z,c.w, d.x,d.y,d.z,d.w};
    v16h r;
#pragma unroll
    for (int i = 0; i < 16; ++i) r[i] = (_Float16)t[i];
    return r;
}

// Build a WMMA A fragment (16 x K32 chunk) from f16 state in LDS (row stride ldh halves).
__device__ __forceinline__ v16h load_afrag_lds(const _Float16* __restrict__ h, int ldh,
                                               int k0, int lane) {
    const int m     = lane & 15;
    const int khalf = (lane & 16) ? 8 : 0;
    const _Float16* p = h + m * ldh + k0 + khalf;
    v8h lo = *(const v8h*)(p);        // k0+khalf+0..7
    v8h hi = *(const v8h*)(p + 16);   // k0+khalf+16..23
    return __builtin_shufflevector(lo, hi, 0,1,2,3,4,5,6,7,8,9,10,11,12,13,14,15);
}

__global__ __launch_bounds__(256)
void lstm2_fc_kernel(const float* __restrict__ x,
                     const float* __restrict__ Wih1, const float* __restrict__ Whh1,
                     const float* __restrict__ bih1, const float* __restrict__ bhh1,
                     const float* __restrict__ Wih2, const float* __restrict__ Whh2,
                     const float* __restrict__ bih2, const float* __restrict__ bhh2,
                     const float* __restrict__ Wfc,  const float* __restrict__ bfc,
                     float* __restrict__ out) {
    __shared__ float      gates1[16 * 256];                 // 16 KB
    __shared__ float      gates2[16 * 128];                 // 8 KB
    __shared__ float      c1s[16 * H1];                     // 4 KB
    __shared__ float      c2s[16 * H2];                     // 2 KB
    __shared__ __align__(16) _Float16 h1h[16 * H1LD];       // f16 h1 state
    __shared__ __align__(16) _Float16 h2h[16 * H2LD];       // f16 h2 state
    __shared__ float      xts[32];                          // x_t tile [16][2]

    const int tid  = threadIdx.x;
    const int wave = tid >> 5;            // 0..7
    const int lane = tid & 31;
    const int ln16 = lane & 15;
    const int mb   = (lane >> 4) * 8;     // C/D: M = v + mb
    const int b0   = blockIdx.x * 16;     // batch tile base

    // ---- Resident weight fragments (constant over all 4096 steps) ----
    // Layer 1: wave owns gate columns [32*wave, 32*wave+32) -> tiles 2w, 2w+1.
    v16h b1f[2][2];
    float bsum1[2], wx0_[2], wx1_[2];
#pragma unroll
    for (int ti = 0; ti < 2; ++ti) {
        const int col = 32 * wave + 16 * ti + ln16;
#pragma unroll
        for (int kc = 0; kc < 2; ++kc)
            b1f[ti][kc] = load_bfrag_f32(Whh1, H1, col, kc * 32, lane);
        bsum1[ti] = bih1[col] + bhh1[col];
        wx0_[ti]  = Wih1[col * 2 + 0];
        wx1_[ti]  = Wih1[col * 2 + 1];
    }
    // Layer 2: wave owns gate columns [16*wave, 16*wave+16) -> tile w.
    const int col2 = 16 * wave + ln16;
    v16h b2i[2], b2h;
#pragma unroll
    for (int kc = 0; kc < 2; ++kc)
        b2i[kc] = load_bfrag_f32(Wih2, H1, col2, kc * 32, lane);
    b2h = load_bfrag_f32(Whh2, H2, col2, 0, lane);
    const float bsum2 = bih2[col2] + bhh2[col2];

    // ---- Zero-init state; preload x_0 ----
    for (int i = tid; i < 16 * H1LD; i += 256) h1h[i] = (_Float16)0.f;
    for (int i = tid; i < 16 * H2LD; i += 256) h2h[i] = (_Float16)0.f;
    for (int i = tid; i < 16 * H1;   i += 256) c1s[i] = 0.f;
    for (int i = tid; i < 16 * H2;   i += 256) c2s[i] = 0.f;
    if (tid < 32) {
        const int b = tid >> 1, ch = tid & 1;
        xts[tid] = x[(size_t)(b0 + b) * T_STEPS * 2 + ch];
    }

#pragma unroll 1
    for (int t = 0; t < T_STEPS; ++t) {
        __syncthreads();   // xts / h2h ready

        // ===== Layer 1: gates1 = x_t@Wih1^T + h1@Whh1^T + b =====
        v16h a1lo = load_afrag_lds(h1h, H1LD, 0,  lane);
        v16h a1hi = load_afrag_lds(h1h, H1LD, 32, lane);
#pragma unroll
        for (int ti = 0; ti < 2; ++ti) {
            v8f acc;
#pragma unroll
            for (int v = 0; v < 8; ++v) {
                const int m = mb + v;
                acc[v] = bsum1[ti] + xts[m * 2] * wx0_[ti] + xts[m * 2 + 1] * wx1_[ti];
            }
            acc = __builtin_amdgcn_wmma_f32_16x16x32_f16(false, a1lo, false, b1f[ti][0],
                                                         (short)0, acc, false, false);
            acc = __builtin_amdgcn_wmma_f32_16x16x32_f16(false, a1hi, false, b1f[ti][1],
                                                         (short)0, acc, false, false);
            const int col = 32 * wave + 16 * ti + ln16;
#pragma unroll
            for (int v = 0; v < 8; ++v) gates1[(mb + v) * 256 + col] = acc[v];
        }
        __syncthreads();   // gates1 complete

        // ===== Layer 1 elementwise: c1,h1 update (1024 elems / 256 threads) =====
#pragma unroll
        for (int j = 0; j < 4; ++j) {
            const int idx = tid + j * 256;
            const int r = idx >> 6, k = idx & 63;
            const float gi = sigmoidf_fast(gates1[r * 256 + k]);
            const float gf = sigmoidf_fast(gates1[r * 256 + 64 + k]);
            const float gg = tanhf_fast   (gates1[r * 256 + 128 + k]);
            const float go = sigmoidf_fast(gates1[r * 256 + 192 + k]);
            const float c  = gf * c1s[r * 64 + k] + gi * gg;
            c1s[r * 64 + k]   = c;
            h1h[r * H1LD + k] = (_Float16)(go * tanhf_fast(c));
        }
        __syncthreads();   // h1h updated

        // ===== Layer 2: gates2 = h1@Wih2^T + h2@Whh2^T + b =====
        a1lo = load_afrag_lds(h1h, H1LD, 0,  lane);
        a1hi = load_afrag_lds(h1h, H1LD, 32, lane);
        v16h a2 = load_afrag_lds(h2h, H2LD, 0, lane);
        {
            v8f acc;
#pragma unroll
            for (int v = 0; v < 8; ++v) acc[v] = bsum2;
            acc = __builtin_amdgcn_wmma_f32_16x16x32_f16(false, a1lo, false, b2i[0],
                                                         (short)0, acc, false, false);
            acc = __builtin_amdgcn_wmma_f32_16x16x32_f16(false, a1hi, false, b2i[1],
                                                         (short)0, acc, false, false);
            acc = __builtin_amdgcn_wmma_f32_16x16x32_f16(false, a2,   false, b2h,
                                                         (short)0, acc, false, false);
#pragma unroll
            for (int v = 0; v < 8; ++v) gates2[(mb + v) * 128 + col2] = acc[v];
        }
        __syncthreads();   // gates2 complete

        // ===== Layer 2 elementwise (512 elems) + prefetch next x tile =====
#pragma unroll
        for (int j = 0; j < 2; ++j) {
            const int idx = tid + j * 256;
            const int r = idx >> 5, k = idx & 31;
            const float gi = sigmoidf_fast(gates2[r * 128 + k]);
            const float gf = sigmoidf_fast(gates2[r * 128 + 32 + k]);
            const float gg = tanhf_fast   (gates2[r * 128 + 64 + k]);
            const float go = sigmoidf_fast(gates2[r * 128 + 96 + k]);
            const float c  = gf * c2s[r * 32 + k] + gi * gg;
            c2s[r * 32 + k]   = c;
            h2h[r * H2LD + k] = (_Float16)(go * tanhf_fast(c));
        }
        if (tid < 32) {
            const int b = tid >> 1, ch = tid & 1;
            const size_t base = (size_t)(b0 + b) * T_STEPS * 2;
            if (t + 1 < T_STEPS) xts[tid] = x[base + (size_t)(t + 1) * 2 + ch];
            if (t + 16 < T_STEPS) __builtin_prefetch(&x[base + (size_t)(t + 16) * 2 + ch], 0, 1);
        }
    }
    __syncthreads();   // final h2h ready

    // ===== FC: out = h2_last @ Wfc^T + bfc  (one 16x16x32 WMMA by wave 0) =====
    if (tid < 32) {
        v16h a2 = load_afrag_lds(h2h, H2LD, 0, lane);
        v16h bf = load_bfrag_f32(Wfc, H2, ln16, 0, lane);
        v8f acc;
#pragma unroll
        for (int v = 0; v < 8; ++v) acc[v] = bfc[ln16];
        acc = __builtin_amdgcn_wmma_f32_16x16x32_f16(false, a2, false, bf,
                                                     (short)0, acc, false, false);
#pragma unroll
        for (int v = 0; v < 8; ++v) out[(size_t)(b0 + mb + v) * LAT + ln16] = acc[v];
    }
}

extern "C" void kernel_launch(void* const* d_in, const int* in_sizes, int n_in,
                              void* d_out, int out_size, void* d_ws, size_t ws_size,
                              hipStream_t stream) {
    (void)in_sizes; (void)n_in; (void)out_size; (void)d_ws; (void)ws_size;
    const float* x    = (const float*)d_in[0];
    const float* Wih1 = (const float*)d_in[1];
    const float* Whh1 = (const float*)d_in[2];
    const float* bih1 = (const float*)d_in[3];
    const float* bhh1 = (const float*)d_in[4];
    const float* Wih2 = (const float*)d_in[5];
    const float* Whh2 = (const float*)d_in[6];
    const float* bih2 = (const float*)d_in[7];
    const float* bhh2 = (const float*)d_in[8];
    const float* Wfc  = (const float*)d_in[9];
    const float* bfc  = (const float*)d_in[10];
    float* out = (float*)d_out;

    dim3 grid(BATCH / 16);   // 32 workgroups, one 16-row batch tile each
    dim3 block(256);         // 8 wave32s
    lstm2_fc_kernel<<<grid, block, 0, stream>>>(x, Wih1, Whh1, bih1, bhh1,
                                                Wih2, Whh2, bih2, bhh2,
                                                Wfc, bfc, out);
}
